// SetNetwork_68298569941674
// MI455X (gfx1250) — matrix operand order repaired and checked by reading, not compiled, not run on hardware
//
#include <hip/hip_runtime.h>

// ---------------------------------------------------------------------------
// SetNetwork forward for MI455X (gfx1250, wave32, WMMA).
// Heavy path: per-request MLP (x_req@W_in -> relu -> @W_e1 -> relu -> masked
// sum-pool) done with v_wmma_f32_16x16x32_f16 on f16-converted operands,
// skipping request tiles beyond n_req. Each wave owns two output column
// tiles whose WMMA chains are interleaved for matrix-pipe ILP and to hide
// the WMMA->VALU hazard slots. Tail layers in fp32 VALU.
// ---------------------------------------------------------------------------

typedef __attribute__((ext_vector_type(16))) _Float16 v16h;
typedef __attribute__((ext_vector_type(8)))  _Float16 h8;
typedef __attribute__((ext_vector_type(8)))  float    v8f;

#define BATCH    2048
#define MAXR     512
#define REQ_IN   64
#define REQ_HID  256
#define EMB1     256
#define EMB2     128
#define INST_IN  64
#define CHID     512

// LDS row strides (in halves), padded (+8) to avoid bank conflicts
#define SW_IN   72     // W_in^T  [256 n][72]
#define SW_E1   264    // W_e1^T  [256 n][264]
#define SA1     72     // A tile  [16 m][72]
#define SH1     264    // h1 tile [16 m][264]

// byte offsets inside dynamic LDS (all 16B aligned)
#define OFF_WIN   0
#define OFF_WE1   (OFF_WIN + 256*SW_IN*2)        // 36864
#define OFF_A1    (OFF_WE1 + 256*SW_E1*2)        // 172032
#define OFF_H1    (OFF_A1  + 16*SA1*2)           // 174336
#define OFF_POOL  (OFF_H1  + 16*SH1*2)           // 182784
#define OFF_CAT   (OFF_POOL + 256*4)             // 183808
#define OFF_HID   (OFF_CAT  + 192*4)             // 184576
#define OFF_PART  (OFF_HID  + 512*4)             // 186624
#define LDS_BYTES (OFF_PART + 256*4)             // 187648 (< 320KB WGP LDS)

// A-matrix fragment (16x32 f16, ISA 7.12.2): lanes 0-15 hold rows m=lane with
// halves = K{0..7,16..23}; lanes 16-31 same rows, K{8..15,24..31}.
__device__ __forceinline__ v16h load_a_frag(const _Float16* tile, int stride,
                                            int lane, int k0) {
  const int m  = lane & 15;
  const int ko = (lane < 16) ? 0 : 8;
  const _Float16* p = tile + m * stride + k0 + ko;
  h8 lo = *(const h8*)(p);
  h8 hi = *(const h8*)(p + 16);
  v16h a;
#pragma unroll
  for (int i = 0; i < 8; ++i) { a[i] = lo[i]; a[i + 8] = hi[i]; }
  return a;
}

// B-matrix fragment (32x16 f16): lane holds column n=lane&15; lanes 0-15 carry
// K{0..7,16..23}, lanes 16-31 carry K{8..15,24..31} (same order as A; weights
// staged transposed: row n, contiguous k).
__device__ __forceinline__ v16h load_b_frag(const _Float16* rowp, int lane, int k0) {
  const int ko = (lane < 16) ? 0 : 8;
  const _Float16* p = rowp + k0 + ko;
  h8 lo = *(const h8*)(p);
  h8 hi = *(const h8*)(p + 16);
  v16h b;
#pragma unroll
  for (int i = 0; i < 8; ++i) { b[i] = lo[i]; b[i + 8] = hi[i]; }
  return b;
}

__global__ __launch_bounds__(256, 1)
void setnet_fwd_kernel(const float* __restrict__ x_inst,
                       const float* __restrict__ x_req,
                       const int*   __restrict__ x_n_req,
                       const float* __restrict__ W_in,
                       const float* __restrict__ W_e1,
                       const float* __restrict__ W_e2,
                       const float* __restrict__ W_c,
                       const float* __restrict__ b_c,
                       const float* __restrict__ W_o,
                       const float* __restrict__ b_o,
                       float* __restrict__ out) {
  extern __shared__ __align__(16) unsigned char smem[];
  _Float16* WinT = (_Float16*)(smem + OFF_WIN);
  _Float16* We1T = (_Float16*)(smem + OFF_WE1);
  _Float16* A1   = (_Float16*)(smem + OFF_A1);
  _Float16* H1   = (_Float16*)(smem + OFF_H1);
  float* pooled  = (float*)(smem + OFF_POOL);
  float* catv    = (float*)(smem + OFF_CAT);
  float* hid     = (float*)(smem + OFF_HID);
  float* part    = (float*)(smem + OFF_PART);

  const int b    = blockIdx.x;
  const int tid  = threadIdx.x;
  const int lane = tid & 31;
  const int wv   = tid >> 5;               // wave id 0..7

  // ---- stage transposed f16 weights into LDS (hits L2 after first blocks) --
#pragma unroll 4
  for (int idx = tid; idx < REQ_IN * REQ_HID; idx += 256) {
    const int k = idx >> 8, n = idx & 255;     // W_in[k][n], k<64, n<256
    WinT[n * SW_IN + k] = (_Float16)W_in[idx];
  }
#pragma unroll 4
  for (int idx = tid; idx < REQ_HID * EMB1; idx += 256) {
    const int k = idx >> 8, n = idx & 255;     // W_e1[k][n]
    We1T[n * SW_E1 + k] = (_Float16)W_e1[idx];
  }

  const int nreq   = x_n_req[b];
  const int ntiles = (nreq + 15) >> 4;         // skip fully-padded tiles

  const int nloc  = lane & 15;
  const int mhalf = (lane < 16) ? 0 : 8;
  const int n0 = wv * 16 + nloc;               // this wave's two output columns
  const int n1 = (wv + 8) * 16 + nloc;

  float accPool0 = 0.f, accPool1 = 0.f;        // masked sum-pool accumulators

  __syncthreads();

  for (int t = 0; t < ntiles; ++t) {
    const int r0 = t << 4;
    // ---- stage A tile: 16 rows x 64 fp32 -> f16 LDS (1 float4 per thread) --
    {
      const int row = tid >> 4;
      const int c4  = (tid & 15) << 2;
      const float4 ld = *(const float4*)(x_req +
          (((size_t)b * MAXR) + r0 + row) * REQ_IN + c4);
      _Float16* dst = A1 + row * SA1 + c4;
      dst[0] = (_Float16)ld.x; dst[1] = (_Float16)ld.y;
      dst[2] = (_Float16)ld.z; dst[3] = (_Float16)ld.w;
    }
    __syncthreads();

    // ---- GEMM1: [16x64] @ [64x256] -> relu -> H1 (f16) --------------------
    // Two interleaved accumulator chains (independent WMMAs) sharing A.
    {
      v8f c0 = {}, c1 = {};
#pragma unroll
      for (int kk = 0; kk < 2; ++kk) {
        v16h a  = load_a_frag(A1, SA1, lane, kk * 32);
        v16h b0 = load_b_frag(WinT + n0 * SW_IN, lane, kk * 32);
        v16h b1 = load_b_frag(WinT + n1 * SW_IN, lane, kk * 32);
        c0 = __builtin_amdgcn_wmma_f32_16x16x32_f16(false, a, false, b0,
                                                    (short)0, c0, false, false);
        c1 = __builtin_amdgcn_wmma_f32_16x16x32_f16(false, a, false, b1,
                                                    (short)0, c1, false, false);
      }
#pragma unroll
      for (int v = 0; v < 8; ++v) {
        H1[(v + mhalf) * SH1 + n0] = (_Float16)fmaxf(c0[v], 0.f);
        H1[(v + mhalf) * SH1 + n1] = (_Float16)fmaxf(c1[v], 0.f);
      }
    }
    __syncthreads();

    // ---- GEMM2: [16x256] @ [256x256] -> relu -> masked row-sum ------------
    {
      v8f c0 = {}, c1 = {};
#pragma unroll
      for (int kk = 0; kk < 8; ++kk) {
        v16h a  = load_a_frag(H1, SH1, lane, kk * 32);
        v16h b0 = load_b_frag(We1T + n0 * SW_E1, lane, kk * 32);
        v16h b1 = load_b_frag(We1T + n1 * SW_E1, lane, kk * 32);
        c0 = __builtin_amdgcn_wmma_f32_16x16x32_f16(false, a, false, b0,
                                                    (short)0, c0, false, false);
        c1 = __builtin_amdgcn_wmma_f32_16x16x32_f16(false, a, false, b1,
                                                    (short)0, c1, false, false);
      }
      float s0 = 0.f, s1 = 0.f;
#pragma unroll
      for (int v = 0; v < 8; ++v) {
        const bool valid = (r0 + v + mhalf) < nreq;    // ragged mask
        s0 += valid ? fmaxf(c0[v], 0.f) : 0.f;         // relu + mask
        s1 += valid ? fmaxf(c1[v], 0.f) : 0.f;
      }
      accPool0 += s0;
      accPool1 += s1;
    }
    __syncthreads();
  }

  // lanes L and L^16 own the same column (m=0..7 vs m=8..15): combine, write
  {
    const float o0 = __shfl_xor(accPool0, 16, 32);
    const float o1 = __shfl_xor(accPool1, 16, 32);
    const float t0 = accPool0 + o0;
    const float t1 = accPool1 + o1;
    if (lane < 16) {
      pooled[wv * 16 + lane]       = t0;
      pooled[(wv + 8) * 16 + lane] = t1;
    }
  }
  __syncthreads();

  // ---- tail (tiny, fp32): req_embed -> concat -> hidden -> scalar out -----
  if (tid < INST_IN) catv[tid] = x_inst[(size_t)b * INST_IN + tid];
  if (tid >= 64 && tid < 64 + EMB2) {
    const int j = tid - 64;
    float s = 0.f;
    for (int k = 0; k < EMB1; ++k) s = fmaf(pooled[k], W_e2[k * EMB2 + j], s);
    catv[64 + j] = fmaxf(s, 0.f);
  }
  __syncthreads();

  {
    float s0 = b_c[tid];
    float s1 = b_c[tid + 256];
    for (int i = 0; i < INST_IN + EMB2; ++i) {
      const float ci = catv[i];
      s0 = fmaf(ci, W_c[i * CHID + tid], s0);
      s1 = fmaf(ci, W_c[i * CHID + tid + 256], s1);
    }
    hid[tid]       = fmaxf(s0, 0.f);
    hid[tid + 256] = fmaxf(s1, 0.f);
  }
  __syncthreads();

  part[tid] = hid[tid] * W_o[tid] + hid[tid + 256] * W_o[tid + 256];
  __syncthreads();
#pragma unroll
  for (int s = 128; s > 0; s >>= 1) {
    if (tid < s) part[tid] += part[tid + s];
    __syncthreads();
  }
  if (tid == 0) out[b] = part[0] + b_o[0];
}

extern "C" void kernel_launch(void* const* d_in, const int* in_sizes, int n_in,
                              void* d_out, int out_size, void* d_ws, size_t ws_size,
                              hipStream_t stream) {
  (void)in_sizes; (void)n_in; (void)out_size; (void)d_ws; (void)ws_size;
  const float* x_inst  = (const float*)d_in[0];
  const float* x_req   = (const float*)d_in[1];
  const int*   x_n_req = (const int*)  d_in[2];
  const float* W_in    = (const float*)d_in[3];
  const float* W_e1    = (const float*)d_in[4];
  const float* W_e2    = (const float*)d_in[5];
  const float* W_c     = (const float*)d_in[6];
  const float* b_c     = (const float*)d_in[7];
  const float* W_o     = (const float*)d_in[8];
  const float* b_o     = (const float*)d_in[9];
  float* out = (float*)d_out;

  dim3 grid(BATCH), block(256);
  hipLaunchKernelGGL(setnet_fwd_kernel, grid, block, LDS_BYTES, stream,
                     x_inst, x_req, x_n_req, W_in, W_e1, W_e2, W_c, b_c,
                     W_o, b_o, out);
}
